// SMARTAgentDecoder_74388833567301
// MI455X (gfx1250) — matrix operand order repaired and checked by reading, compiled, not run on hardware
//
#include <hip/hip_runtime.h>
#include <hip/hip_bf16.h>

typedef unsigned short bf16_t;
typedef __attribute__((ext_vector_type(16))) __bf16 v16bf;
typedef __attribute__((ext_vector_type(8)))  __bf16 v8bf;
typedef __attribute__((ext_vector_type(8)))  float  v8f;

#define PI_F     3.14159265358979323846f
#define TWOPI_F  6.28318530717958647692f

// ---------------- device helpers ----------------
__device__ __forceinline__ bf16_t f2b(float f) {
  unsigned u = __float_as_uint(f);
  unsigned r = (u + 0x7FFFu + ((u >> 16) & 1u)) >> 16;   // RNE
  return (bf16_t)r;
}
__device__ __forceinline__ unsigned encf(float f) {      // order-preserving f32->u32
  unsigned u = __float_as_uint(f);
  return (u & 0x80000000u) ? ~u : (u | 0x80000000u);
}
__device__ __forceinline__ float decf(unsigned u) {
  return (u & 0x80000000u) ? __uint_as_float(u ^ 0x80000000u) : __uint_as_float(~u);
}
__device__ __forceinline__ float wrapang(float a) {
  float t = a + PI_F;
  t = t - floorf(t * (1.0f / TWOPI_F)) * TWOPI_F;
  return t - PI_F;
}

// ---------------- WMMA GEMM ----------------
// C[MxNc] = A[MxK](bf16 row-major, K % 32 == 0) * B(bf16, fragment-swizzled) (+bias)(+acc)(+fuse)
// Each wave computes a 16x64 strip (4 N-tiles, A fragment reused 4x). 4 waves/block over M.
// B swizzled layout: [( (k0/32)*(Nc/16) + n0/16 ) * 512 + lane*16 + i], value = W[k][n] with
//   k = k0 + (lane>>4)*16 + i, n = n0 + (lane&15)   (zero for k >= K_src)
__global__ __launch_bounds__(128)
void k_gemm(const bf16_t* __restrict__ A, const bf16_t* __restrict__ B,
            const float* __restrict__ bias, float* __restrict__ C,
            bf16_t* __restrict__ Cb, int M, int K, int Nc, int fuse, int accf)
{
  const int n0 = blockIdx.x * 64;
  const int m0 = (blockIdx.y * 4 + threadIdx.y) * 16;
  if (m0 >= M) return;
  const int lane = threadIdx.x;
  const int half = lane >> 4;
  const int mr = lane & 15;
  int mA = m0 + mr; if (mA > M - 1) mA = M - 1;   // rows >= M never stored; clamp is safe
  const int tiles_n = Nc >> 4;
  const bf16_t* __restrict__ Arow = A + (size_t)mA * K;

  v8f a0 = {0.f,0.f,0.f,0.f,0.f,0.f,0.f,0.f};
  v8f a1 = a0, a2 = a0, a3 = a0;

  for (int k0 = 0; k0 < K; k0 += 32) {
    // A fragment: one contiguous 16-elem run k = k0 + half*16 + i (matches B swizzle)
    v8bf alo = *(const v8bf*)(Arow + k0 + half * 16);
    v8bf ahi = *(const v8bf*)(Arow + k0 + half * 16 + 8);
    v16bf af = __builtin_shufflevector(alo, ahi, 0,1,2,3,4,5,6,7,8,9,10,11,12,13,14,15);
    const bf16_t* __restrict__ Bt =
        B + ((size_t)(k0 >> 5) * tiles_n + (n0 >> 4)) * 512 + lane * 16;
    v16bf b0 = *(const v16bf*)(Bt);
    v16bf b1 = *(const v16bf*)(Bt + 512);
    v16bf b2 = *(const v16bf*)(Bt + 1024);
    v16bf b3 = *(const v16bf*)(Bt + 1536);
    a0 = __builtin_amdgcn_wmma_f32_16x16x32_bf16(false, af, false, b0, (short)0, a0, false, false);
    a1 = __builtin_amdgcn_wmma_f32_16x16x32_bf16(false, af, false, b1, (short)0, a1, false, false);
    a2 = __builtin_amdgcn_wmma_f32_16x16x32_bf16(false, af, false, b2, (short)0, a2, false, false);
    a3 = __builtin_amdgcn_wmma_f32_16x16x32_bf16(false, af, false, b3, (short)0, a3, false, false);
  }

#pragma unroll
  for (int t = 0; t < 4; ++t) {
    const v8f* accp = (t == 0) ? &a0 : (t == 1) ? &a1 : (t == 2) ? &a2 : &a3;
    const int n = n0 + t * 16 + mr;
    const float bv = bias ? bias[n] : 0.0f;
#pragma unroll
    for (int r = 0; r < 8; ++r) {
      const int m = m0 + half * 8 + r;   // C/D layout: VGPR r -> row r (+8 upper half-wave)
      if (m < M) {
        float v = (*accp)[r] + bv;
        if (accf) v += C[(size_t)m * Nc + n];
        if (fuse == 1)      v = v > 0.f ? v : 0.f;
        else if (fuse == 2) v = 1.0f / (1.0f + expf(-v));
        if (C)  C[(size_t)m * Nc + n]  = v;
        if (Cb) Cb[(size_t)m * Nc + n] = f2b(v);
      }
    }
  }
}

// weight f32 -> bf16 fragment-swizzle (zero-pads k in [K, K32))
__global__ void k_cvt_sw(const float* __restrict__ W, bf16_t* __restrict__ out,
                         int K, int Nc, size_t total) {
  size_t idx = blockIdx.x * (size_t)blockDim.x + threadIdx.x;
  if (idx >= total) return;
  size_t tile = idx >> 9;
  int within = (int)(idx & 511);
  int lane = within >> 4, i = within & 15;
  int tiles_n = Nc >> 4;
  int kt = (int)(tile / tiles_n), nt = (int)(tile % tiles_n);
  int k = kt * 32 + (lane >> 4) * 16 + i;
  int n = nt * 16 + (lane & 15);
  out[idx] = (k < K) ? f2b(W[(size_t)k * Nc + n]) : (bf16_t)0;
}

// activation f32 -> bf16 with column zero-padding (Ksrc -> Kdst)
__global__ void k_cvt_pad(const float* __restrict__ src, bf16_t* __restrict__ dst,
                          int M, int Ksrc, int Kdst) {
  size_t i = blockIdx.x * (size_t)blockDim.x + threadIdx.x;
  if (i >= (size_t)M * Kdst) return;
  int m = (int)(i / Kdst), k = (int)(i % Kdst);
  dst[i] = (k < Ksrc) ? f2b(src[(size_t)m * Ksrc + k]) : (bf16_t)0;
}

// ---------------- LayerNorm over D=128 (wave per row) ----------------
__global__ void k_ln128(const float* __restrict__ X, const float* __restrict__ g,
                        const float* __restrict__ b, const float* __restrict__ res,
                        float* __restrict__ outF, bf16_t* __restrict__ outB,
                        int M, int relu)
{
  int row = blockIdx.x * blockDim.y + threadIdx.y;
  if (row >= M) return;
  int lane = threadIdx.x;
  const float* xr = X + (size_t)row * 128;
  float v[4]; float s = 0.f, sq = 0.f;
#pragma unroll
  for (int t = 0; t < 4; ++t) { v[t] = xr[lane + 32 * t]; s += v[t]; sq += v[t] * v[t]; }
#pragma unroll
  for (int o = 16; o > 0; o >>= 1) { s += __shfl_xor(s, o, 32); sq += __shfl_xor(sq, o, 32); }
  float mean = s * (1.0f / 128.0f);
  float var  = sq * (1.0f / 128.0f) - mean * mean;
  var = var < 0.f ? 0.f : var;
  float rstd = rsqrtf(var + 1e-5f);
#pragma unroll
  for (int t = 0; t < 4; ++t) {
    int c = lane + 32 * t;
    float y = (v[t] - mean) * rstd * g[c] + b[c];
    if (relu) y = y > 0.f ? y : 0.f;
    if (res)  y += res[(size_t)row * 128 + c];
    if (outF) outF[(size_t)row * 128 + c] = y;
    if (outB) outB[(size_t)row * 128 + c] = f2b(y);
  }
}

// ---------------- small elementwise kernels ----------------
__global__ void k_cvt(const float* __restrict__ s, bf16_t* __restrict__ d, size_t n) {
  size_t i = blockIdx.x * (size_t)blockDim.x + threadIdx.x;
  if (i < n) d[i] = f2b(s[i]);
}
__global__ void k_fillf(float* p, float v, size_t n) {
  size_t i = blockIdx.x * (size_t)blockDim.x + threadIdx.x;
  if (i < n) p[i] = v;
}
__global__ void k_fillu(unsigned* p, unsigned v, size_t n) {
  size_t i = blockIdx.x * (size_t)blockDim.x + threadIdx.x;
  if (i < n) p[i] = v;
}
__global__ void k_add_catrep(float* __restrict__ sum, const float* __restrict__ cat, int M, int rep) {
  size_t i = blockIdx.x * (size_t)blockDim.x + threadIdx.x;
  if (i >= (size_t)M * 128) return;
  int m = (int)(i >> 7), c = (int)(i & 127);
  sum[i] += cat[(size_t)(m / rep) * 128 + c];
}
__global__ void k_add_typeemb(float* __restrict__ cat, const float* __restrict__ temb,
                              const int* __restrict__ atype, int N) {
  size_t i = blockIdx.x * (size_t)blockDim.x + threadIdx.x;
  if (i >= (size_t)N * 128) return;
  int nidx = (int)(i >> 7), c = (int)(i & 127);
  cat[i] += temb[(size_t)atype[nidx] * 128 + c];
}
__global__ void k_motion(const float* __restrict__ pos_a, const float* __restrict__ head_a,
                         float* __restrict__ feat, int N, int S) {
  int idx = blockIdx.x * blockDim.x + threadIdx.x;
  if (idx >= N * S) return;
  int s = idx % S;
  float mx = 0.f, my = 0.f;
  if (s > 0) {
    mx = pos_a[(size_t)idx * 2]     - pos_a[(size_t)(idx - 1) * 2];
    my = pos_a[(size_t)idx * 2 + 1] - pos_a[(size_t)(idx - 1) * 2 + 1];
  }
  float sq = mx * mx + my * my;
  float h = head_a[idx], cx = cosf(h), cy = sinf(h);
  float mnorm, mang;
  if (sq == 0.f) { mnorm = 0.f; mang = 0.f; }
  else { mnorm = sqrtf(sq); mang = atan2f(cx * my - cy * mx, cx * mx + cy * my); }
  feat[(size_t)idx * 2] = mnorm; feat[(size_t)idx * 2 + 1] = mang;
}
__global__ void k_edge_t(const int* __restrict__ ei, const float* __restrict__ pos_a,
                         const float* __restrict__ head_a, float* __restrict__ r, int E) {
  int e = blockIdx.x * blockDim.x + threadIdx.x;
  if (e >= E) return;
  int s0 = ei[e], s1 = ei[E + e];
  float rx = pos_a[(size_t)s0 * 2]     - pos_a[(size_t)s1 * 2];
  float ry = pos_a[(size_t)s0 * 2 + 1] - pos_a[(size_t)s1 * 2 + 1];
  float h0 = head_a[s0], h1 = head_a[s1];
  float cx = cosf(h1), cy = sinf(h1);
  r[(size_t)e * 4 + 0] = sqrtf(rx * rx + ry * ry);
  r[(size_t)e * 4 + 1] = atan2f(cx * ry - cy * rx, cx * rx + cy * ry);
  r[(size_t)e * 4 + 2] = wrapang(h0 - h1);
  r[(size_t)e * 4 + 3] = (float)(s0 - s1);
}
__global__ void k_edge_a2a(const int* __restrict__ ei, const float* __restrict__ pos_a,
                           const float* __restrict__ head_a, float* __restrict__ r,
                           int E, int N, int S) {
  int e = blockIdx.x * blockDim.x + threadIdx.x;
  if (e >= E) return;
  int a0 = ei[e], a1 = ei[E + e];
  int n0 = a0 % N, t0 = a0 / N, n1 = a1 % N, t1 = a1 / N;
  size_t i0 = (size_t)(n0 * S + t0), i1 = (size_t)(n1 * S + t1);
  float rx = pos_a[i0 * 2]     - pos_a[i1 * 2];
  float ry = pos_a[i0 * 2 + 1] - pos_a[i1 * 2 + 1];
  float h0 = head_a[i0], h1 = head_a[i1];
  float cx = cosf(h1), cy = sinf(h1);
  r[(size_t)e * 3 + 0] = sqrtf(rx * rx + ry * ry);
  r[(size_t)e * 3 + 1] = atan2f(cx * ry - cy * rx, cx * rx + cy * ry);
  r[(size_t)e * 3 + 2] = wrapang(h0 - h1);
}
__global__ void k_edge_pl(const int* __restrict__ ei, const float* __restrict__ pos_pl,
                          const float* __restrict__ orient_pl, const float* __restrict__ pos_a,
                          const float* __restrict__ head_a, float* __restrict__ r,
                          int E, int N, int S, int P) {
  int e = blockIdx.x * blockDim.x + threadIdx.x;
  if (e >= E) return;
  int p0 = ei[e], a1 = ei[E + e];
  int p = p0 % P, n1 = a1 % N, t1 = a1 / N;
  size_t i1 = (size_t)(n1 * S + t1);
  float rx = pos_pl[(size_t)p * 2]     - pos_a[i1 * 2];
  float ry = pos_pl[(size_t)p * 2 + 1] - pos_a[i1 * 2 + 1];
  float h1 = head_a[i1];
  float cx = cosf(h1), cy = sinf(h1);
  r[(size_t)e * 3 + 0] = sqrtf(rx * rx + ry * ry);
  r[(size_t)e * 3 + 1] = atan2f(cx * ry - cy * rx, cx * rx + cy * ry);
  r[(size_t)e * 3 + 2] = wrapang(orient_pl[p] - h1);
}
// fourier features, padded to 160 columns: [cos(64) | sin(64) | x | 0-pad(31)]
__global__ void k_fourier_feat(const float* __restrict__ x, const float* __restrict__ freqs,
                               bf16_t* __restrict__ out, int M, int IC, int c) {
  size_t i = blockIdx.x * (size_t)blockDim.x + threadIdx.x;
  if (i >= (size_t)M * 160) return;
  int m = (int)(i / 160), col = (int)(i % 160);
  float xv = x[(size_t)m * IC + c];
  float val;
  if (col < 64)        val = cosf(TWOPI_F * xv * freqs[c * 64 + col]);
  else if (col < 128)  val = sinf(TWOPI_F * xv * freqs[c * 64 + col - 64]);
  else if (col == 128) val = xv;
  else                 val = 0.f;
  out[i] = f2b(val);
}
__global__ void k_fusion_in(const float* __restrict__ ev, const float* __restrict__ ep,
                            const float* __restrict__ ec, const int* __restrict__ tok_idx,
                            const int* __restrict__ atype, const float* __restrict__ x_a,
                            bf16_t* __restrict__ out, int N, int S) {
  size_t i = blockIdx.x * (size_t)blockDim.x + threadIdx.x;
  if (i >= (size_t)N * S * 128) return;
  int m = (int)(i >> 7), c = (int)(i & 127);
  int n = m / S;
  int ty = atype[n], ti = tok_idx[m];
  const float* e = (ty == 0) ? ev : ((ty == 1) ? ep : ec);
  out[(size_t)m * 256 + c]       = f2b(e[(size_t)ti * 128 + c]);
  out[(size_t)m * 256 + 128 + c] = f2b(x_a[(size_t)m * 128 + c]);
}
__global__ void k_transpose(const float* __restrict__ in, float* __restrict__ out,
                            int N, int S, int dir) {
  size_t i = blockIdx.x * (size_t)blockDim.x + threadIdx.x;
  if (i >= (size_t)N * S * 128) return;
  int m = (int)(i >> 7), c = (int)(i & 127);
  int om;
  if (dir == 0) { int n = m / S, s = m % S; om = s * N + n; }  // NS -> SN
  else          { int s = m / N, n = m % N; om = n * S + s; }  // SN -> NS
  out[(size_t)om * 128 + c] = in[i];
}
__global__ void k_gate_combine(const float* __restrict__ agg, const float* __restrict__ g,
                               const float* __restrict__ sb, bf16_t* __restrict__ msg, size_t n) {
  size_t i = blockIdx.x * (size_t)blockDim.x + threadIdx.x;
  if (i < n) { float a = agg[i]; msg[i] = f2b(a + g[i] * (sb[i] - a)); }
}

// ---------------- edge attention passes ----------------
__global__ void k_attn_sim(const int* __restrict__ ei, int E, int srcmod,
                           const float* __restrict__ q, const float* __restrict__ k,
                           const float* __restrict__ kr, float* __restrict__ sim,
                           unsigned* __restrict__ mEnc) {
  size_t t = blockIdx.x * (size_t)blockDim.x + threadIdx.x;
  if (t >= (size_t)E * 8) return;
  int e = (int)(t >> 3), h = (int)(t & 7);
  int src = ei[e], dst = ei[E + e];
  int ks = srcmod ? (src % srcmod) : src;
  const float* qp  = q  + (size_t)dst * 128 + h * 16;
  const float* kp  = k  + (size_t)ks  * 128 + h * 16;
  const float* krp = kr + (size_t)e   * 128 + h * 16;
  float s = 0.f;
#pragma unroll
  for (int d = 0; d < 16; ++d) s += qp[d] * (kp[d] + krp[d]);
  s *= 0.25f;  // DH^-0.5
  sim[t] = s;
  atomicMax(&mEnc[(size_t)dst * 8 + h], encf(s));
}
__global__ void k_attn_ez(const int* __restrict__ ei, int E, const unsigned* __restrict__ mEnc,
                          float* __restrict__ sim, float* __restrict__ z) {
  size_t t = blockIdx.x * (size_t)blockDim.x + threadIdx.x;
  if (t >= (size_t)E * 8) return;
  int e = (int)(t >> 3), h = (int)(t & 7);
  int dst = ei[E + e];
  float m = decf(mEnc[(size_t)dst * 8 + h]);
  float w = expf(sim[t] - m);
  sim[t] = w;
  atomicAdd(&z[(size_t)dst * 8 + h], w);
}
__global__ void k_attn_agg(const int* __restrict__ ei, int E, int srcmod,
                           const float* __restrict__ v, const float* __restrict__ vr,
                           const float* __restrict__ sim, const float* __restrict__ z,
                           float* __restrict__ agg) {
  size_t t = blockIdx.x * (size_t)blockDim.x + threadIdx.x;
  if (t >= (size_t)E * 128) return;
  int e = (int)(t >> 7), c = (int)(t & 127), h = c >> 4;
  int src = ei[e], dst = ei[E + e];
  int ks = srcmod ? (src % srcmod) : src;
  float w = sim[(size_t)e * 8 + h] / z[(size_t)dst * 8 + h];
  atomicAdd(&agg[(size_t)dst * 128 + c],
            (v[(size_t)ks * 128 + c] + vr[(size_t)e * 128 + c]) * w);
}

// ---------------- host ----------------
static inline unsigned nblk(size_t n, unsigned t) { return (unsigned)((n + t - 1) / t); }

extern "C" void kernel_launch(void* const* d_in, const int* in_sizes, int n_in,
                              void* d_out, int out_size, void* d_ws, size_t ws_size,
                              hipStream_t stream)
{
  (void)ws_size; (void)out_size;
  if (n_in < 615) return;
  const int N = 512, S = 18, P = 2048, NS = N * S, NTOK = 2048;

  const float* pos_a     = (const float*)d_in[0];
  const float* head_a    = (const float*)d_in[1];
  const int*   tok_idx   = (const int*)d_in[2];
  const int*   atype     = (const int*)d_in[3];
  const float* ashape    = (const float*)d_in[4];
  const float* tveh      = (const float*)d_in[5];
  const float* tped      = (const float*)d_in[6];
  const float* tcyc      = (const float*)d_in[7];
  const float* pt_token  = (const float*)d_in[8];
  const float* pos_pl    = (const float*)d_in[9];
  const float* orient_pl = (const float*)d_in[10];
  const int*   ei_t      = (const int*)d_in[11];
  const int*   ei_a2a    = (const int*)d_in[12];
  const int*   ei_pl     = (const int*)d_in[13];
  const int Et = in_sizes[11] / 2, Ea = in_sizes[12] / 2, Ep = in_sizes[13] / 2;

  auto PF = [&](int i) -> const float* { return (const float*)d_in[i]; };

  // bump allocator over workspace
  char* wsp = (char*)d_ws; size_t off = 0;
  auto alloc  = [&](size_t b) -> void* { size_t a = (off + 255) & ~(size_t)255; off = a + b; return wsp + a; };
  auto allocF = [&](size_t n) { return (float*)alloc(n * 4); };
  auto allocB = [&](size_t n) { return (bf16_t*)alloc(n * 2); };
  // weight -> swizzled bf16 fragments
  auto cvtW = [&](const float* W, int K, int Nc) -> bf16_t* {
    int K32 = (K + 31) & ~31;
    size_t total = (size_t)K32 * Nc;
    bf16_t* d = allocB(total);
    k_cvt_sw<<<nblk(total, 256), 256, 0, stream>>>(W, d, K, Nc, total);
    return d;
  };
  auto gemm = [&](const bf16_t* A, const bf16_t* B, const float* bias, float* C, bf16_t* Cb,
                  int M, int K, int Nc, int fuse, int accf) {
    dim3 g((unsigned)(Nc / 64), nblk((size_t)M, 64));
    dim3 blk(32, 4);
    k_gemm<<<g, blk, 0, stream>>>(A, B, bias, C, Cb, M, K, Nc, fuse, accf);
  };
  auto ln = [&](const float* X, const float* g, const float* b, const float* res,
                float* oF, bf16_t* oB, int M, int relu) {
    dim3 blk(32, 8);
    k_ln128<<<nblk((size_t)M, 8), blk, 0, stream>>>(X, g, b, res, oF, oB, M, relu);
  };

  // ---- convert weights (param leaves in setup_inputs() dict insertion order) ----
  bf16_t* shW1 = cvtW(PF(15), 3, 128);          // padded to K=32
  bf16_t* shW2 = cvtW(PF(19), 128, 128);

  struct FourierW { const float* freqs; bf16_t* W1; const float* b1; const float* lng; const float* lnb;
                    bf16_t* W2; const float* b2; const float* olg; const float* olb;
                    bf16_t* Wout; const float* bout; int IC; };
  auto prepFourier = [&](int base, int IC) -> FourierW {
    FourierW f; f.IC = IC; f.freqs = PF(base);
    f.W1 = allocB((size_t)IC * 160 * 128);      // per-channel swizzled (K=129 -> 160)
    for (int c = 0; c < IC; ++c)
      k_cvt_sw<<<nblk((size_t)160 * 128, 256), 256, 0, stream>>>(
          PF(base + 1) + (size_t)c * 129 * 128, f.W1 + (size_t)c * 160 * 128, 129, 128,
          (size_t)160 * 128);
    f.b1 = PF(base + 2); f.lng = PF(base + 3); f.lnb = PF(base + 4);
    f.W2 = allocB((size_t)IC * 128 * 128);
    for (int c = 0; c < IC; ++c)
      k_cvt_sw<<<nblk((size_t)128 * 128, 256), 256, 0, stream>>>(
          PF(base + 5) + (size_t)c * 128 * 128, f.W2 + (size_t)c * 128 * 128, 128, 128,
          (size_t)128 * 128);
    f.b2 = PF(base + 6); f.olg = PF(base + 7); f.olb = PF(base + 8);
    f.Wout = cvtW(PF(base + 9), 128, 128); f.bout = PF(base + 10);
    return f;
  };
  FourierW fxa = prepFourier(21, 2), frt = prepFourier(32, 4),
           frpl = prepFourier(43, 3), fra = prepFourier(54, 3);

  struct EmbW { bf16_t *W1, *W2, *W3; const float *b1, *g1, *bb1, *b2, *g2, *bb2, *b3; int Kp; };
  auto prepEmb = [&](int base, int K) -> EmbW {
    EmbW e; e.Kp = (K + 31) & ~31;
    e.W1 = cvtW(PF(base), K, 128);       e.b1 = PF(base + 1); e.g1 = PF(base + 2); e.bb1 = PF(base + 3);
    e.W2 = cvtW(PF(base + 4), 128, 128); e.b2 = PF(base + 5); e.g2 = PF(base + 6); e.bb2 = PF(base + 7);
    e.W3 = cvtW(PF(base + 8), 128, 128); e.b3 = PF(base + 9);
    return e;
  };
  EmbW evh = prepEmb(65, 8), eped = prepEmb(75, 8), ecyc = prepEmb(85, 8), efus = prepEmb(95, 256);

  struct AttnW { bf16_t *Wq, *Wk, *Wv, *Wkr, *Wvr, *Ws, *Wg1, *Wg2, *Wo, *Wf1, *Wf2;
                 const float *bq, *bs, *bg, *bo, *bf1, *bf2;
                 const float *lsg, *lsb, *ldg, *ldb, *lrg, *lrb, *pg, *pb, *fpg, *fpb, *fqg, *fqb; };
  auto prepAttn = [&](int base) -> AttnW {
    AttnW a;
    a.Wq  = cvtW(PF(base + 0), 128, 128);  a.bq  = PF(base + 1);
    a.Wk  = cvtW(PF(base + 2), 128, 128);
    a.Wv  = cvtW(PF(base + 3), 128, 128);
    a.Wkr = cvtW(PF(base + 4), 128, 128);
    a.Wvr = cvtW(PF(base + 5), 128, 128);
    a.Ws  = cvtW(PF(base + 6), 128, 128);  a.bs  = PF(base + 7);
    a.Wg1 = cvtW(PF(base + 8), 128, 128);                       // Wg rows 0..127 (agg)
    a.Wg2 = cvtW(PF(base + 8) + 128 * 128, 128, 128);           // Wg rows 128..255 (xd)
    a.bg  = PF(base + 9);
    a.Wo  = cvtW(PF(base + 10), 128, 128); a.bo  = PF(base + 11);
    a.Wf1 = cvtW(PF(base + 12), 128, 512); a.bf1 = PF(base + 13);
    a.Wf2 = cvtW(PF(base + 14), 512, 128); a.bf2 = PF(base + 15);
    a.lsg = PF(base + 16); a.lsb = PF(base + 17); a.ldg = PF(base + 18); a.ldb = PF(base + 19);
    a.lrg = PF(base + 20); a.lrb = PF(base + 21); a.pg  = PF(base + 22); a.pb  = PF(base + 23);
    a.fpg = PF(base + 24); a.fpb = PF(base + 25); a.fqg = PF(base + 26); a.fqb = PF(base + 27);
    return a;
  };
  AttnW At[6], Ap[6], Aa[6];
  for (int i = 0; i < 6; ++i) {
    At[i] = prepAttn(105 + i * 28);
    Ap[i] = prepAttn(273 + i * 28);
    Aa[i] = prepAttn(441 + i * 28);
  }
  bf16_t* hW1 = cvtW(PF(609), 128, 128);
  const float *hb1 = PF(610), *hlg = PF(611), *hlb = PF(612);
  bf16_t* hW2 = cvtW(PF(613), 128, 2048);
  const float* hb2 = PF(614);

  // ---- shared scratch ----
  int maxE = Et; if (Ea > maxE) maxE = Ea; if (Ep > maxE) maxE = Ep;
  int maxM = maxE; if (NS > maxM) maxM = NS;
  bf16_t* tA160 = allocB((size_t)maxM * 160);
  float*  tF1   = allocF((size_t)maxM * 128);
  bf16_t* tB1   = allocB((size_t)maxM * 128);
  float*  tSum  = allocF((size_t)maxM * 128);

  auto fourier = [&](const FourierW& f, int M, const float* x, const float* cat, int rep, float* out) {
    for (int c = 0; c < f.IC; ++c) {
      k_fourier_feat<<<nblk((size_t)M * 160, 256), 256, 0, stream>>>(x, f.freqs, tA160, M, f.IC, c);
      gemm(tA160, f.W1 + (size_t)c * 160 * 128, f.b1 + c * 128, tF1, nullptr, M, 160, 128, 0, 0);
      ln(tF1, f.lng + c * 128, f.lnb + c * 128, nullptr, nullptr, tB1, M, 1);
      gemm(tB1, f.W2 + (size_t)c * 128 * 128, f.b2 + c * 128, tSum, nullptr, M, 128, 128, 0, (c > 0) ? 1 : 0);
    }
    if (cat) k_add_catrep<<<nblk((size_t)M * 128, 256), 256, 0, stream>>>(tSum, cat, M, rep);
    ln(tSum, f.olg, f.olb, nullptr, nullptr, tB1, M, 1);
    gemm(tB1, f.Wout, f.bout, out, nullptr, M, 128, 128, 0, 0);
  };
  auto mlp_emb = [&](const EmbW& e, const bf16_t* Xb, int M, float* out) {
    gemm(Xb, e.W1, e.b1, tF1, nullptr, M, e.Kp, 128, 0, 0);
    ln(tF1, e.g1, e.bb1, nullptr, nullptr, tB1, M, 1);
    gemm(tB1, e.W2, e.b2, tF1, nullptr, M, 128, 128, 0, 0);
    ln(tF1, e.g2, e.bb2, nullptr, nullptr, tB1, M, 1);
    gemm(tB1, e.W3, e.b3, out, nullptr, M, 128, 128, 0, 0);
  };

  // ---- token embeddings (K padded 8 -> 32) ----
  float* embV = allocF((size_t)NTOK * 128);
  float* embP = allocF((size_t)NTOK * 128);
  float* embC = allocF((size_t)NTOK * 128);
  {
    bf16_t* tb = allocB((size_t)NTOK * 32);
    k_cvt_pad<<<nblk((size_t)NTOK * 32, 256), 256, 0, stream>>>(tveh, tb, NTOK, 8, 32);
    mlp_emb(evh, tb, NTOK, embV);
    k_cvt_pad<<<nblk((size_t)NTOK * 32, 256), 256, 0, stream>>>(tped, tb, NTOK, 8, 32);
    mlp_emb(eped, tb, NTOK, embP);
    k_cvt_pad<<<nblk((size_t)NTOK * 32, 256), 256, 0, stream>>>(tcyc, tb, NTOK, 8, 32);
    mlp_emb(ecyc, tb, NTOK, embC);
  }

  // ---- cat = type_a_emb[type] + shape MLP (K padded 3 -> 32) ----
  float* cat = allocF((size_t)N * 128);
  {
    bf16_t* shb = allocB((size_t)N * 32);
    k_cvt_pad<<<nblk((size_t)N * 32, 256), 256, 0, stream>>>(ashape, shb, N, 3, 32);
    gemm(shb, shW1, PF(16), tF1, nullptr, N, 32, 128, 0, 0);
    ln(tF1, PF(17), PF(18), nullptr, nullptr, tB1, N, 1);
    gemm(tB1, shW2, PF(20), cat, nullptr, N, 128, 128, 0, 0);
    k_add_typeemb<<<nblk((size_t)N * 128, 256), 256, 0, stream>>>(cat, PF(14), atype, N);
  }

  // ---- motion features + x_a fourier ----
  float* feat = allocF((size_t)NS * 2);
  k_motion<<<nblk((size_t)NS, 256), 256, 0, stream>>>(pos_a, head_a, feat, N, S);
  float* x_a = allocF((size_t)NS * 128);
  fourier(fxa, NS, feat, cat, S, x_a);

  // ---- fusion ----
  bf16_t* fin = allocB((size_t)NS * 256);
  k_fusion_in<<<nblk((size_t)NS * 128, 256), 256, 0, stream>>>(embV, embP, embC, tok_idx, atype, x_a, fin, N, S);
  float* x_ns = allocF((size_t)NS * 128);
  mlp_emb(efus, fin, NS, x_ns);

  // ---- edge relative features + fourier embeddings ----
  float* rt = allocF((size_t)Et * 4);
  k_edge_t<<<nblk((size_t)Et, 256), 256, 0, stream>>>(ei_t, pos_a, head_a, rt, Et);
  float* rtE = allocF((size_t)Et * 128);
  fourier(frt, Et, rt, nullptr, 1, rtE);
  float* ra = allocF((size_t)Ea * 3);
  k_edge_a2a<<<nblk((size_t)Ea, 256), 256, 0, stream>>>(ei_a2a, pos_a, head_a, ra, Ea, N, S);
  float* raE = allocF((size_t)Ea * 128);
  fourier(fra, Ea, ra, nullptr, 1, raE);
  float* rp = allocF((size_t)Ep * 3);
  k_edge_pl<<<nblk((size_t)Ep, 256), 256, 0, stream>>>(ei_pl, pos_pl, orient_pl, pos_a, head_a, rp, Ep, N, S, P);
  float* rpE = allocF((size_t)Ep * 128);
  fourier(frpl, Ep, rp, nullptr, 1, rpE);

  // ---- attention scratch ----
  float*  x_sn = allocF((size_t)NS * 128);
  bf16_t* xsB  = allocB((size_t)NS * 128);
  bf16_t* xdB  = allocB((size_t)NS * 128);
  bf16_t* xplB = allocB((size_t)P * 128);
  float*  qbuf = allocF((size_t)NS * 128);
  float*  kbuf = allocF((size_t)NS * 128);
  float*  vbuf = allocF((size_t)NS * 128);
  bf16_t* rnB  = allocB((size_t)maxE * 128);
  float*  krvr = allocF((size_t)maxE * 128);
  float*  simb = allocF((size_t)maxE * 8);
  unsigned* mEnc = (unsigned*)alloc((size_t)NS * 8 * 4);
  float*  zbuf = allocF((size_t)NS * 8);
  float*  aggb = allocF((size_t)NS * 128);
  bf16_t* aggB = allocB((size_t)NS * 128);
  float*  gpre = allocF((size_t)NS * 128);
  float*  sbuf = allocF((size_t)NS * 128);
  bf16_t* msgB = allocB((size_t)NS * 128);
  float*  ffh  = allocF((size_t)NS * 512);
  bf16_t* ffhB = allocB((size_t)NS * 512);
  float*  tmpO = allocF((size_t)NS * 128);

  auto attn = [&](const AttnW& w, float* x, int M, const float* rE, int E, const int* ei,
                  int bipartite, const float* srcx, int srcM, int srcmod) {
    const bf16_t* xd;
    if (!bipartite) {
      ln(x, w.lsg, w.lsb, nullptr, nullptr, xsB, M, 0);
      gemm(xsB, w.Wq, w.bq, qbuf, nullptr, M, 128, 128, 0, 0);
      gemm(xsB, w.Wk, nullptr, kbuf, nullptr, M, 128, 128, 0, 0);
      gemm(xsB, w.Wv, nullptr, vbuf, nullptr, M, 128, 128, 0, 0);
      xd = xsB;
    } else {
      ln(srcx, w.lsg, w.lsb, nullptr, nullptr, xplB, srcM, 0);
      gemm(xplB, w.Wk, nullptr, kbuf, nullptr, srcM, 128, 128, 0, 0);
      gemm(xplB, w.Wv, nullptr, vbuf, nullptr, srcM, 128, 128, 0, 0);
      ln(x, w.ldg, w.ldb, nullptr, nullptr, xdB, M, 0);
      gemm(xdB, w.Wq, w.bq, qbuf, nullptr, M, 128, 128, 0, 0);
      xd = xdB;
    }
    ln(rE, w.lrg, w.lrb, nullptr, nullptr, rnB, E, 0);
    gemm(rnB, w.Wkr, nullptr, krvr, nullptr, E, 128, 128, 0, 0);     // kr
    k_fillu<<<nblk((size_t)M * 8, 256), 256, 0, stream>>>(mEnc, 0u, (size_t)M * 8);
    k_fillf<<<nblk((size_t)M * 8, 256), 256, 0, stream>>>(zbuf, 0.f, (size_t)M * 8);
    k_fillf<<<nblk((size_t)M * 128, 256), 256, 0, stream>>>(aggb, 0.f, (size_t)M * 128);
    k_attn_sim<<<nblk((size_t)E * 8, 256), 256, 0, stream>>>(ei, E, srcmod, qbuf, kbuf, krvr, simb, mEnc);
    k_attn_ez<<<nblk((size_t)E * 8, 256), 256, 0, stream>>>(ei, E, mEnc, simb, zbuf);
    gemm(rnB, w.Wvr, nullptr, krvr, nullptr, E, 128, 128, 0, 0);     // vr (reuse)
    k_attn_agg<<<nblk((size_t)E * 128, 256), 256, 0, stream>>>(ei, E, srcmod, vbuf, krvr, simb, zbuf, aggb);
    // gate + skip
    k_cvt<<<nblk((size_t)M * 128, 256), 256, 0, stream>>>(aggb, aggB, (size_t)M * 128);
    gemm(aggB, w.Wg1, w.bg, gpre, nullptr, M, 128, 128, 0, 0);
    gemm(xd, w.Wg2, nullptr, gpre, nullptr, M, 128, 128, 2, 1);      // += then sigmoid
    gemm(xd, w.Ws, w.bs, sbuf, nullptr, M, 128, 128, 0, 0);
    k_gate_combine<<<nblk((size_t)M * 128, 256), 256, 0, stream>>>(aggb, gpre, sbuf, msgB, (size_t)M * 128);
    gemm(msgB, w.Wo, w.bo, tmpO, nullptr, M, 128, 128, 0, 0);
    ln(tmpO, w.pg, w.pb, x, x, nullptr, M, 0);                        // x += LN(msg@Wo+bo)
    // FFN
    ln(x, w.fpg, w.fpb, nullptr, nullptr, tB1, M, 0);
    gemm(tB1, w.Wf1, w.bf1, ffh, ffhB, M, 128, 512, 1, 0);            // relu
    gemm(ffhB, w.Wf2, w.bf2, tmpO, nullptr, M, 512, 128, 0, 0);
    ln(tmpO, w.fqg, w.fqb, x, x, nullptr, M, 0);                      // x += LN(ff)
  };

  // ---- main loop ----
  for (int l = 0; l < 6; ++l) {
    attn(At[l], x_ns, NS, rtE, Et, ei_t, 0, nullptr, 0, 0);
    k_transpose<<<nblk((size_t)NS * 128, 256), 256, 0, stream>>>(x_ns, x_sn, N, S, 0);
    attn(Ap[l], x_sn, NS, rpE, Ep, ei_pl, 1, pt_token, P, P);
    attn(Aa[l], x_sn, NS, raE, Ea, ei_a2a, 0, nullptr, 0, 0);
    k_transpose<<<nblk((size_t)NS * 128, 256), 256, 0, stream>>>(x_sn, x_ns, N, S, 1);
  }

  // ---- head: relu(LN(x@W1+b1)) @ W2 + b2 -> (N*S, 2048) ----
  k_cvt<<<nblk((size_t)NS * 128, 256), 256, 0, stream>>>(x_ns, xsB, (size_t)NS * 128);
  gemm(xsB, hW1, hb1, tF1, nullptr, NS, 128, 128, 0, 0);
  ln(tF1, hlg, hlb, nullptr, nullptr, tB1, NS, 1);
  gemm(tB1, hW2, hb2, (float*)d_out, nullptr, NS, 128, 2048, 0, 0);
}
